// QbfSimpleEncoder_24507083391178
// MI455X (gfx1250) — matrix operand order repaired and checked by reading, not compiled, not run on hardware
//
#include <hip/hip_runtime.h>
#include <hip/hip_bf16.h>

// ---------------------------------------------------------------------------
// QBF simple encoder for MI455X (gfx1250, wave32, WMMA).
//
// Roofline: the two 16384x4096 clause matrices dominate traffic. Convert them
// ONCE to bf16 (row-major + transposed row-major) so every big GEMM pass
// streams 128MB instead of 256MB and runs on v_wmma_f32_16x16x32_bf16.
// Big GEMMs are K-split for occupancy with a deterministic partial reduction.
// B tiles are staged global->LDS with GLOBAL_LOAD_ASYNC_TO_LDS_B128
// (ASYNCcnt, double-buffered) to kill the 4x redundant strided B fetch.
// ---------------------------------------------------------------------------

#define NV 4096
#define NC 16384
#define VE 128
#define KPAD 160   // 136 padded to a multiple of 32
#define ITERS 4
#define BROW 136   // LDS B row stride in elements (128 + 8 pad -> bank spread)

typedef __attribute__((ext_vector_type(16))) __bf16 v16bf;
typedef __attribute__((ext_vector_type(8)))  __bf16 v8bf;
typedef __attribute__((ext_vector_type(8)))  float  v8f;

union Frag { v16bf v; v8bf h[2]; };

// --- fragment loaders (row-major bf16 sources, 16B-aligned chunks) ----------
// A 16x32 bf16 frag: lane<16 -> M=lane,   K in {0..7, 16..23}
//                    lane>=16-> M=lane-16,K in {8..15,24..31}
__device__ inline Frag load_a(const __bf16* __restrict__ A, int lda,
                              int row0, int k0, int lane) {
  int half = lane >> 4, r = lane & 15;
  const __bf16* p = A + (size_t)(row0 + r) * lda + k0 + half * 8;
  Frag f;
  f.h[0] = *(const v8bf*)(p);
  f.h[1] = *(const v8bf*)(p + 16);
  return f;
}
// B 32x16 bf16 frag: lane<16 -> N=lane, K in {0..7,16..23}; lane>=16 -> N=lane-16, K in {8..15,24..31}
__device__ inline Frag load_b(const __bf16* __restrict__ B, int ldb,
                              int k0, int n0, int lane) {
  int half = lane >> 4, n = lane & 15;
  const __bf16* p = B + (size_t)(k0 + half * 8) * ldb + n0 + n;
  Frag f;
#pragma unroll
  for (int s = 0; s < 8; ++s) {
    f.v[s] = p[(size_t)s * ldb];
    f.v[8 + s] = p[(size_t)(16 + s) * ldb];
  }
  return f;
}
// Same B frag but from an LDS-staged 32xBROW tile (strided column reads).
__device__ inline Frag load_b_lds(const __bf16* base, int n0, int lane) {
  int half = lane >> 4, n = lane & 15;
  const __bf16* p = base + half * 8 * BROW + n0 + n;
  Frag f;
#pragma unroll
  for (int s = 0; s < 8; ++s) {
    f.v[s] = p[s * BROW];
    f.v[8 + s] = p[(16 + s) * BROW];
  }
  return f;
}

__device__ inline v8f wmma_bf16(const Frag& a, const Frag& b, v8f c) {
  return __builtin_amdgcn_wmma_f32_16x16x32_bf16(false, a.v, false, b.v,
                                                 (short)0, c, false, false);
}

// --- CDNA5 async global->LDS copy (tracked by ASYNCcnt) ---------------------
// vdst VGPR = LDS byte offset; flat shared pointers carry it in addr[31:0].
__device__ inline void async_load_b128(void* lds_ptr, const void* gptr) {
  unsigned lds_off = (unsigned)(size_t)lds_ptr;
  unsigned long long ga = (unsigned long long)(size_t)gptr;
  asm volatile("global_load_async_to_lds_b128 %0, %1, off"
               :: "v"(lds_off), "v"(ga) : "memory");
}
__device__ inline void wait_async_le4() {
  asm volatile("s_wait_asynccnt 0x4" ::: "memory");
}
__device__ inline void wait_async_0() {
  asm volatile("s_wait_asynccnt 0x0" ::: "memory");
}

// --- one-time prep ----------------------------------------------------------

// fp32 [136][128] -> bf16 [160][128] with zero padding rows
__global__ __launch_bounds__(256) void k_cvt_w(const float* __restrict__ w,
                                               __bf16* __restrict__ o) {
  int i = blockIdx.x * 256 + threadIdx.x;
  if (i >= KPAD * VE) return;
  int r = i >> 7, c = i & 127;
  float v = (r < 136) ? w[r * VE + c] : 0.0f;
  o[i] = (__bf16)v;
}

// vs[NV][160]: cols 0..7 = vlabels, rest zero (initial vembs = 0)
__global__ __launch_bounds__(256) void k_build_vs(const float* __restrict__ vl,
                                                  __bf16* __restrict__ vs) {
  int i = blockIdx.x * 256 + threadIdx.x;  // NV*160
  int r = i / KPAD, c = i - r * KPAD;
  float v = (c < 8) ? vl[r * 8 + c] : 0.0f;
  vs[i] = (__bf16)v;
}

// cs[NC][160]: cols 0..7 = clabels, cols 136..159 zero (8..135 by GEMM1)
__global__ __launch_bounds__(256) void k_build_cs(const float* __restrict__ cl,
                                                  __bf16* __restrict__ cs) {
  int i = blockIdx.x * 256 + threadIdx.x;  // NC*32
  int r = i >> 5, j = i & 31;
  int c = (j < 8) ? j : (128 + j);
  float v = (c < 8) ? cl[r * 8 + c] : 0.0f;
  cs[(size_t)r * KPAD + c] = (__bf16)v;
}

// cmat fp32 -> bf16 row-major + bf16 transposed row-major (tiled LDS)
__global__ __launch_bounds__(256) void k_cvt_cmat(const float* __restrict__ cp,
                                                  const float* __restrict__ cn,
                                                  __bf16* __restrict__ op,
                                                  __bf16* __restrict__ on,
                                                  __bf16* __restrict__ otp,
                                                  __bf16* __restrict__ otn) {
  const float* src = blockIdx.z ? cn : cp;
  __bf16* orm = blockIdx.z ? on : op;
  __bf16* otm = blockIdx.z ? otn : otp;
  int c0 = blockIdx.x * 32;  // NV dim
  int r0 = blockIdx.y * 32;  // NC dim
  __shared__ __bf16 t[32][33];
  int tid = threadIdx.x;
#pragma unroll
  for (int i = 0; i < 4; ++i) {
    int idx = tid + i * 256;
    int r = idx >> 5, c = idx & 31;
    __bf16 b = (__bf16)src[(size_t)(r0 + r) * NV + (c0 + c)];
    orm[(size_t)(r0 + r) * NV + (c0 + c)] = b;
    t[r][c] = b;
  }
  __syncthreads();
#pragma unroll
  for (int i = 0; i < 4; ++i) {
    int idx = tid + i * 256;
    int r = idx >> 5, c = idx & 31;
    otm[(size_t)(c0 + r) * NC + (r0 + c)] = t[c][r];
  }
}

// --- small GEMM: out{p,n}[M][128] = A[M][160] @ W{p,n}[160][128] ------------
__global__ __launch_bounds__(256) void k_small_gemm(
    const __bf16* __restrict__ A, const __bf16* __restrict__ Wp,
    const __bf16* __restrict__ Wn, __bf16* __restrict__ outp,
    __bf16* __restrict__ outn) {
  int wave = threadIdx.x >> 5, lane = threadIdx.x & 31;
  int wid = blockIdx.x * 8 + wave;
  int mt = wid >> 3, nt = wid & 7;
  v8f cp = {}, cn = {};
#pragma unroll
  for (int ks = 0; ks < KPAD / 32; ++ks) {
    int k0 = ks * 32;
    Frag a = load_a(A, KPAD, mt * 16, k0, lane);
    Frag bp = load_b(Wp, VE, k0, nt * 16, lane);
    Frag bn = load_b(Wn, VE, k0, nt * 16, lane);
    cp = wmma_bf16(a, bp, cp);
    cn = wmma_bf16(a, bn, cn);
  }
  int half = lane >> 4;
  int col = nt * 16 + (lane & 15);
  int row = mt * 16 + 8 * half;
#pragma unroll
  for (int r = 0; r < 8; ++r) {
    outp[(size_t)(row + r) * VE + col] = (__bf16)cp[r];
    outn[(size_t)(row + r) * VE + col] = (__bf16)cn[r];
  }
}

// --- big GEMM: Pacc[ksplit][M][128] += Ap@Bp + An@Bn (per K-chunk) ----------
// block = 8 waves on a 128(M) x 128(N) tile; wave = 32x64 sub-tile.
// B tiles (32x128 per matrix) staged to LDS via async copy, double-buffered;
// each wave issues exactly 4 async loads per stage -> s_wait_asynccnt 4.
__global__ __launch_bounds__(256) void k_big_gemm(
    const __bf16* __restrict__ Ap, const __bf16* __restrict__ An,
    const __bf16* __restrict__ Bp, const __bf16* __restrict__ Bn,
    float* __restrict__ Pacc, int M, int K, int KC) {
  __shared__ __align__(16) __bf16 Bsh[2][2][32][BROW];
  int tid = threadIdx.x;
  int wave = tid >> 5, lane = tid & 31;
  int wm = wave >> 1, wn = wave & 1;
  int row0 = blockIdx.x * 128 + wm * 32;
  int nbase = wn * 64;
  int kbeg = blockIdx.y * KC, kend = kbeg + KC;

  // cooperative async stage of one 32x128 B tile pair into buffer s
  auto stage_B = [&](int s, int kk) {
#pragma unroll
    for (int u = 0; u < 2; ++u) {
      int c = tid + u * 256;          // 512 16B-chunks per matrix tile
      int row = c >> 4, col16 = c & 15;
      const __bf16* gp = Bp + (size_t)(kk + row) * VE + col16 * 8;
      const __bf16* gn = Bn + (size_t)(kk + row) * VE + col16 * 8;
      async_load_b128(&Bsh[s][0][row][col16 * 8], gp);
      async_load_b128(&Bsh[s][1][row][col16 * 8], gn);
    }
  };

  v8f zf = {};
  v8f acc[2][4];
#pragma unroll
  for (int i = 0; i < 2; ++i)
#pragma unroll
    for (int j = 0; j < 4; ++j) acc[i][j] = zf;

  stage_B(0, kbeg);
  int s = 0;
  for (int kk = kbeg; kk < kend; kk += 32) {
    bool more = (kk + 32) < kend;
    if (more) stage_B(s ^ 1, kk + 32);  // prefetch next stage (async)
    if (more) wait_async_le4(); else wait_async_0();
    __syncthreads();                    // stage s visible to all waves

    Frag ap[2], an[2];
#pragma unroll
    for (int i = 0; i < 2; ++i) {
      ap[i] = load_a(Ap, K, row0 + i * 16, kk, lane);
      an[i] = load_a(An, K, row0 + i * 16, kk, lane);
    }
    if (more) {  // prefetch next A k-slice (global_prefetch_b8)
      __builtin_prefetch(Ap + (size_t)(row0 + (lane & 15)) * K + kk + 32, 0, 1);
      __builtin_prefetch(An + (size_t)(row0 + (lane & 15)) * K + kk + 32, 0, 1);
    }
#pragma unroll
    for (int j = 0; j < 4; ++j) {
      Frag bp = load_b_lds(&Bsh[s][0][0][0], nbase + j * 16, lane);
      Frag bn = load_b_lds(&Bsh[s][1][0][0], nbase + j * 16, lane);
#pragma unroll
      for (int i = 0; i < 2; ++i) {
        acc[i][j] = wmma_bf16(ap[i], bp, acc[i][j]);
        acc[i][j] = wmma_bf16(an[i], bn, acc[i][j]);
      }
    }
    __syncthreads();  // all waves done reading stage s before it is re-staged
    s ^= 1;
  }

  float* base = Pacc + (size_t)blockIdx.y * M * VE;
  int half = lane >> 4, r16 = lane & 15;
#pragma unroll
  for (int i = 0; i < 2; ++i)
#pragma unroll
    for (int j = 0; j < 4; ++j) {
      int row = row0 + i * 16 + 8 * half;
      int col = nbase + j * 16 + r16;
#pragma unroll
      for (int r = 0; r < 8; ++r)
        base[(size_t)(row + r) * VE + col] = acc[i][j][r];
    }
}

// --- combine: sum K-split partials + bias + relu ----------------------------
__global__ __launch_bounds__(256) void k_combine_clause(
    const float* __restrict__ Pacc, const float* __restrict__ bias,
    __bf16* __restrict__ cs) {
  int i = blockIdx.x * 256 + threadIdx.x;  // NC*128
  int r = i >> 7, c = i & 127;
  float s = bias[c];
#pragma unroll
  for (int k = 0; k < 4; ++k) s += Pacc[(size_t)k * NC * VE + i];
  s = fmaxf(s, 0.0f);
  cs[(size_t)r * KPAD + 8 + c] = (__bf16)s;
}

__global__ __launch_bounds__(256) void k_combine_var(
    const float* __restrict__ Pacc, const float* __restrict__ bias,
    float* __restrict__ vtmp) {
  int i = blockIdx.x * 256 + threadIdx.x;  // NV*128
  int c = i & 127;
  float s = bias[c];
#pragma unroll
  for (int k = 0; k < 16; ++k) s += Pacc[(size_t)k * NV * VE + i];
  vtmp[i] = fmaxf(s, 0.0f);
}

// --- layernorm over VE=128; writes fp32 result + bf16 copy into vs ---------
__global__ __launch_bounds__(128) void k_layernorm(
    const float* __restrict__ vtmp, const float* __restrict__ gamma,
    const float* __restrict__ beta, float* __restrict__ out,
    __bf16* __restrict__ vs) {
  int r = blockIdx.x, c = threadIdx.x;
  float x = vtmp[(size_t)r * VE + c];
  __shared__ float red[128];
  red[c] = x;
  __syncthreads();
  for (int s = 64; s > 0; s >>= 1) {
    if (c < s) red[c] += red[c + s];
    __syncthreads();
  }
  float mu = red[0] * (1.0f / 128.0f);
  __syncthreads();
  float d = x - mu;
  red[c] = d * d;
  __syncthreads();
  for (int s = 64; s > 0; s >>= 1) {
    if (c < s) red[c] += red[c + s];
    __syncthreads();
  }
  float var = red[0] * (1.0f / 128.0f);
  float y = d * rsqrtf(var + 1e-5f) * gamma[c] + beta[c];
  out[(size_t)r * VE + c] = y;
  vs[(size_t)r * KPAD + 8 + c] = (__bf16)y;
}

// ---------------------------------------------------------------------------

extern "C" void kernel_launch(void* const* d_in, const int* in_sizes, int n_in,
                              void* d_out, int out_size, void* d_ws,
                              size_t ws_size, hipStream_t stream) {
  const float* vlabels = (const float*)d_in[0];
  const float* clabels = (const float*)d_in[1];
  const float* cmat_pos = (const float*)d_in[2];
  const float* cmat_neg = (const float*)d_in[3];
  const float* W_V_pos = (const float*)d_in[4];
  const float* W_V_neg = (const float*)d_in[5];
  const float* B_V = (const float*)d_in[6];
  const float* W_C_pos = (const float*)d_in[7];
  const float* W_C_neg = (const float*)d_in[8];
  const float* B_C = (const float*)d_in[9];
  const float* ln_gamma = (const float*)d_in[10];
  const float* ln_beta = (const float*)d_in[11];
  float* out = (float*)d_out;

  char* w = (char*)d_ws;
  auto alloc = [&](size_t bytes) -> char* {
    char* p = w;
    w += (bytes + 255) & ~(size_t)255;
    return p;
  };
  __bf16* cmP = (__bf16*)alloc((size_t)NC * NV * 2);
  __bf16* cmN = (__bf16*)alloc((size_t)NC * NV * 2);
  __bf16* cmTP = (__bf16*)alloc((size_t)NV * NC * 2);
  __bf16* cmTN = (__bf16*)alloc((size_t)NV * NC * 2);
  __bf16* vs = (__bf16*)alloc((size_t)NV * KPAD * 2);
  __bf16* cs = (__bf16*)alloc((size_t)NC * KPAD * 2);
  __bf16* wvP = (__bf16*)alloc(KPAD * VE * 2);
  __bf16* wvN = (__bf16*)alloc(KPAD * VE * 2);
  __bf16* wcP = (__bf16*)alloc(KPAD * VE * 2);
  __bf16* wcN = (__bf16*)alloc(KPAD * VE * 2);
  __bf16* pv = (__bf16*)alloc((size_t)NV * VE * 2);
  __bf16* nv = (__bf16*)alloc((size_t)NV * VE * 2);
  __bf16* pc = (__bf16*)alloc((size_t)NC * VE * 2);
  __bf16* nc = (__bf16*)alloc((size_t)NC * VE * 2);
  float* Pacc = (float*)alloc((size_t)16 * NV * VE * 4);  // == 4*NC*VE*4
  float* vtmp = (float*)alloc((size_t)NV * VE * 4);

  // one-time prep
  k_cvt_w<<<(KPAD * VE + 255) / 256, 256, 0, stream>>>(W_V_pos, wvP);
  k_cvt_w<<<(KPAD * VE + 255) / 256, 256, 0, stream>>>(W_V_neg, wvN);
  k_cvt_w<<<(KPAD * VE + 255) / 256, 256, 0, stream>>>(W_C_pos, wcP);
  k_cvt_w<<<(KPAD * VE + 255) / 256, 256, 0, stream>>>(W_C_neg, wcN);
  k_build_vs<<<(NV * KPAD) / 256, 256, 0, stream>>>(vlabels, vs);
  k_build_cs<<<(NC * 32) / 256, 256, 0, stream>>>(clabels, cs);
  k_cvt_cmat<<<dim3(NV / 32, NC / 32, 2), 256, 0, stream>>>(
      cmat_pos, cmat_neg, cmP, cmN, cmTP, cmTN);

  for (int it = 0; it < ITERS; ++it) {
    // pv/nv = [vlabels||vembs] @ W_V_{pos,neg}
    k_small_gemm<<<(NV / 16), 256, 0, stream>>>(vs, wvP, wvN, pv, nv);
    // cembs = relu(cmat_pos@pv + cmat_neg@nv + B_V) -> cs[:,8:136]
    k_big_gemm<<<dim3(NC / 128, 4), 256, 0, stream>>>(cmP, cmN, pv, nv, Pacc,
                                                      NC, NV, NV / 4);
    k_combine_clause<<<(NC * VE) / 256, 256, 0, stream>>>(Pacc, B_V, cs);
    // pc/nc = [clabels||cembs] @ W_C_{pos,neg}
    k_small_gemm<<<(NC / 16), 256, 0, stream>>>(cs, wcP, wcN, pc, nc);
    // v = relu(cmat_pos.T@pc + cmat_neg.T@nc + B_C)
    k_big_gemm<<<dim3(NV / 128, 16), 256, 0, stream>>>(cmTP, cmTN, pc, nc,
                                                       Pacc, NV, NC, NC / 16);
    k_combine_var<<<(NV * VE) / 256, 256, 0, stream>>>(Pacc, B_C, vtmp);
    // layernorm -> d_out (fp32) and vs[:,8:136] (bf16 for next iter)
    k_layernorm<<<NV, 128, 0, stream>>>(vtmp, ln_gamma, ln_beta, out, vs);
  }
}
